// Intermediate_73632919323320
// MI455X (gfx1250) — 8-bit matrix operand order repaired and checked by reading, compiled, not run on hardware
//
#include <hip/hip_runtime.h>


typedef int  v8i __attribute__((ext_vector_type(8)));
typedef int  v4i __attribute__((ext_vector_type(4)));
typedef int  v2i __attribute__((ext_vector_type(2)));

#define HIDDEN 4096
#define INTER  16384
#define MTOK   4096

__global__ __launch_bounds__(256) void quant_f32_i8(
    const float* __restrict__ x, int* __restrict__ q,
    const float* __restrict__ clip_ptr, long n4)
{
    const float c = fabsf(*clip_ptr);
    const float inv_scale = 127.0f / c;
    long i      = (long)blockIdx.x * blockDim.x + threadIdx.x;
    long stride = (long)gridDim.x * blockDim.x;
    for (; i < n4; i += stride) {
        float4 v = ((const float4*)x)[i];
        int b0 = (int)__builtin_rintf(fminf(fmaxf(v.x, -c), c) * inv_scale) & 0xff;
        int b1 = (int)__builtin_rintf(fminf(fmaxf(v.y, -c), c) * inv_scale) & 0xff;
        int b2 = (int)__builtin_rintf(fminf(fmaxf(v.z, -c), c) * inv_scale) & 0xff;
        int b3 = (int)__builtin_rintf(fminf(fmaxf(v.w, -c), c) * inv_scale) & 0xff;
        q[i] = b0 | (b1 << 8) | (b2 << 16) | (b3 << 24);
    }
}

__global__ __launch_bounds__(256) void gemm_i8_wmma(
    const signed char* __restrict__ Aq,
    const signed char* __restrict__ Bq,
    const float* __restrict__ bias,
    const float* __restrict__ cx_ptr,
    const float* __restrict__ cw_ptr,
    float* __restrict__ out)
{
    const int lane  = threadIdx.x & 31;
    const int wave  = threadIdx.x >> 5;
    const int wm    = wave >> 2;
    const int wn    = wave & 3;
    const int m0    = blockIdx.x * 128 + wm * 64;
    const int n0    = blockIdx.y * 256 + wn * 64;
    const int lhalf = lane >> 4;
    const int l16   = lane & 15;

    const signed char* aBase[4];
#pragma unroll
    for (int mi = 0; mi < 4; ++mi)
        aBase[mi] = Aq + (long)(m0 + mi * 16 + l16) * HIDDEN + lhalf * 16;

    const signed char* bBase[4];
#pragma unroll
    for (int ni = 0; ni < 4; ++ni)
        bBase[ni] = Bq + (long)(n0 + ni * 16 + l16) * HIDDEN + lhalf * 16;

    v8i acc[4][4] = {};

    for (int k0 = 0; k0 < HIDDEN; k0 += 64) {
        v8i a[4], b[4];
#pragma unroll
        for (int mi = 0; mi < 4; ++mi) {
            const signed char* p = aBase[mi] + k0;
            v2i q0 = *(const v2i*)(p);
            v2i q1 = *(const v2i*)(p + 8);
            v2i q2 = *(const v2i*)(p + 32);
            v2i q3 = *(const v2i*)(p + 40);
            v4i lo = __builtin_shufflevector(q0, q1, 0, 1, 2, 3);
            v4i hi = __builtin_shufflevector(q2, q3, 0, 1, 2, 3);
            a[mi]  = __builtin_shufflevector(lo, hi, 0, 1, 2, 3, 4, 5, 6, 7);
        }
#pragma unroll
        for (int ni = 0; ni < 4; ++ni) {
            const signed char* p = bBase[ni] + k0;
            v4i lo = *(const v4i*)(p);
            v4i hi = *(const v4i*)(p + 32);
            b[ni]  = __builtin_shufflevector(lo, hi, 0, 1, 2, 3, 4, 5, 6, 7);
        }
#pragma unroll
        for (int mi = 0; mi < 4; ++mi)
#pragma unroll
            for (int ni = 0; ni < 4; ++ni)
                acc[mi][ni] = __builtin_amdgcn_wmma_i32_16x16x64_iu8(
                     true, a[mi],  true, b[ni],
                    acc[mi][ni],  false,  false);
    }

    const float s = (fabsf(*cx_ptr) * (1.0f / 127.0f)) *
                    (fabsf(*cw_ptr) * (1.0f / 127.0f));
#pragma unroll
    for (int ni = 0; ni < 4; ++ni) {
        const int col = n0 + ni * 16 + l16;
        const float bv = bias[col];
#pragma unroll
        for (int mi = 0; mi < 4; ++mi) {
            const int rowBase = m0 + mi * 16 + lhalf * 8;
#pragma unroll
            for (int r = 0; r < 8; ++r) {
                float v = (float)acc[mi][ni][r] * s + bv;
                out[(long)(rowBase + r) * INTER + col] = fmaxf(v, 0.0f);
            }
        }
    }
}

extern "C" void kernel_launch(void* const* d_in, const int* in_sizes, int n_in,
                              void* d_out, int out_size, void* d_ws, size_t ws_size,
                              hipStream_t stream) {
    const float* x    = (const float*)d_in[0];
    const float* w    = (const float*)d_in[1];
    const float* bias = (const float*)d_in[2];
    const float* cw   = (const float*)d_in[3];
    const float* cx   = (const float*)d_in[4];
    float* out        = (float*)d_out;

    signed char* wq = (signed char*)d_ws;
    signed char* xq = (signed char*)d_ws + (size_t)INTER * HIDDEN;

    quant_f32_i8<<<8192, 256, 0, stream>>>(w, (int*)wq, cw,
                                           (long)INTER * HIDDEN / 4);
    quant_f32_i8<<<4096, 256, 0, stream>>>(x, (int*)xq, cx,
                                           (long)MTOK * HIDDEN / 4);
    dim3 grid(MTOK / 128, INTER / 256);
    gemm_i8_wmma<<<grid, 256, 0, stream>>>(xq, wq, bias, cx, cw, out);
}
